// MultiHeadAttention_12017318494758
// MI455X (gfx1250) — matrix operand order repaired and checked by reading, not compiled, not run on hardware
//
#include <hip/hip_runtime.h>
#include <hip/hip_bf16.h>

// ---------------------------------------------------------------------------
// Problem constants (match reference)
// ---------------------------------------------------------------------------
constexpr int Bn = 4, Tn = 2048, Cn = 2048, Hn = 16, Dn = 128;
constexpr int NQKV  = 3 * Cn;      // 6144
constexpr int Mrows = Bn * Tn;     // 8192

typedef __attribute__((ext_vector_type(16))) __bf16 v16bf;
typedef __attribute__((ext_vector_type(8)))  float  v8f;

// ---------------------------------------------------------------------------
// bf16 helpers (round-to-nearest-even) — only used OUTSIDE hot GEMM loops
// ---------------------------------------------------------------------------
__device__ inline unsigned short f2bf_bits(float f) {
  unsigned u = __builtin_bit_cast(unsigned, f);
  return (unsigned short)((u + 0x7fffu + ((u >> 16) & 1u)) >> 16);
}
__device__ inline float bf2f(unsigned short h) {
  unsigned u = ((unsigned)h) << 16;
  return __builtin_bit_cast(float, u);
}

__device__ inline v8f wmma_bf16(v16bf a, v16bf b, v8f c) {
  // 8 args: (neg_a, A, neg_b, B, c_mod, C, reuse_a, reuse_b)
  return __builtin_amdgcn_wmma_f32_16x16x32_bf16(false, a, false, b, (short)0, c,
                                                 false, false);
}

// ---------------------------------------------------------------------------
// Fragment loaders (CDNA5 ISA layouts, wave32) — bf16 sources only, so
// each is contiguous 16B runs -> pure global_load_b128 in the hot loops.
// A 16x32: lane m=lane&15, hi8=(lane&16)?8:0; elem e<8 -> k=hi8+e,
//          e>=8 -> k=16+hi8+(e-8). Two contiguous 8-element (16B) runs.
// B 32x16: lane n=lane&15, hi8=(lane&16)?8:0; elem e -> k as for A (e<8: hi8+e, else 16+hi8+(e-8)).
//          Column n of B == row n of W (GEMMs are X*W^T): two 16B runs.
// ---------------------------------------------------------------------------
__device__ inline v16bf load_a_bf16(const unsigned short* A, int lda, int row0,
                                    int k0, int lane) {
  const int m = lane & 15, hi8 = (lane & 16) ? 8 : 0;
  const unsigned short* p = A + (size_t)(row0 + m) * lda + k0;
  v16bf a;
#pragma unroll
  for (int half = 0; half < 2; ++half) {
    const unsigned short* q = p + half * 16 + hi8;
#pragma unroll
    for (int j = 0; j < 8; ++j) a[half * 8 + j] = __builtin_bit_cast(__bf16, q[j]);
  }
  return a;
}

__device__ inline v16bf load_bT_bf16(const unsigned short* W, int ldw, int n0,
                                     int k0, int lane) {
  const int n = lane & 15, hi8 = (lane & 16) ? 8 : 0;
  const unsigned short* p = W + (size_t)(n0 + n) * ldw + k0 + hi8;
  v16bf b;
#pragma unroll
  for (int e = 0; e < 16; ++e) b[e] = __builtin_bit_cast(__bf16, p[e + (e & 8)]);
  return b;
}

// 16-lane-group reductions (xor masks <= 8 stay inside each half-wave group)
__device__ inline float group_max16(float v) {
  v = fmaxf(v, __shfl_xor(v, 1, 32));
  v = fmaxf(v, __shfl_xor(v, 2, 32));
  v = fmaxf(v, __shfl_xor(v, 4, 32));
  v = fmaxf(v, __shfl_xor(v, 8, 32));
  return v;
}
__device__ inline float group_sum16(float v) {
  v += __shfl_xor(v, 1, 32);
  v += __shfl_xor(v, 2, 32);
  v += __shfl_xor(v, 4, 32);
  v += __shfl_xor(v, 8, 32);
  return v;
}

// ---------------------------------------------------------------------------
// Kernel 0: bulk f32 -> bf16 conversion (streaming, bandwidth-bound).
// Each thread: 8 floats in (2x b128), 8 bf16 out (1x b128).
// ---------------------------------------------------------------------------
__global__ __launch_bounds__(256) void cvt_bf16_kernel(
    const float* __restrict__ src, unsigned int* __restrict__ dst, int n8) {
  const int i = blockIdx.x * blockDim.x + threadIdx.x;   // one per 8 floats
  if (i >= n8) return;
  const float4 f0 = ((const float4*)src)[i * 2];
  const float4 f1 = ((const float4*)src)[i * 2 + 1];
  uint4 o;
  o.x = (unsigned)f2bf_bits(f0.x) | ((unsigned)f2bf_bits(f0.y) << 16);
  o.y = (unsigned)f2bf_bits(f0.z) | ((unsigned)f2bf_bits(f0.w) << 16);
  o.z = (unsigned)f2bf_bits(f1.x) | ((unsigned)f2bf_bits(f1.y) << 16);
  o.w = (unsigned)f2bf_bits(f1.z) | ((unsigned)f2bf_bits(f1.w) << 16);
  ((uint4*)dst)[i] = o;
}

// ---------------------------------------------------------------------------
// Kernel 1: QKV = x @ w_qkv^T   (8192 x 2048) x (6144 x 2048)^T, all-bf16 in.
// One wave -> 32x64 output tile (2x4 WMMA accumulators; 8 WMMA : 12 b128/iter)
// Epilogue: Q,K -> [B,H,T,D] bf16 ; V -> transposed [B,H,D,T] bf16.
// ---------------------------------------------------------------------------
__global__ __launch_bounds__(256) void qkv_gemm_kernel(
    const unsigned short* __restrict__ x, const unsigned short* __restrict__ w,
    unsigned short* __restrict__ Qb, unsigned short* __restrict__ Kb,
    unsigned short* __restrict__ Vt) {
  const int lane = threadIdx.x & 31;
  const int wave = threadIdx.x >> 5;
  const int gw   = blockIdx.x * 8 + wave;
  const int tilesN = NQKV / 64;                  // 96
  const int tm = (gw / tilesN) * 32;
  const int tn = (gw % tilesN) * 64;

  v8f acc[2][4] = {};
  for (int k0 = 0; k0 < Cn; k0 += 32) {
    v16bf a0 = load_a_bf16(x, Cn, tm,      k0, lane);
    v16bf a1 = load_a_bf16(x, Cn, tm + 16, k0, lane);
#pragma unroll
    for (int ni = 0; ni < 4; ++ni) {
      v16bf b = load_bT_bf16(w, Cn, tn + ni * 16, k0, lane);
      acc[0][ni] = wmma_bf16(a0, b, acc[0][ni]);
      acc[1][ni] = wmma_bf16(a1, b, acc[1][ni]);
    }
  }

  const int nl = lane & 15;
  const int hi = (lane >> 4) & 1;
#pragma unroll
  for (int mi = 0; mi < 2; ++mi)
#pragma unroll
    for (int ni = 0; ni < 4; ++ni)
#pragma unroll
      for (int r = 0; r < 8; ++r) {
        const int m = tm + mi * 16 + r + 8 * hi;   // row in [0, 8192)
        const int n = tn + ni * 16 + nl;           // col in [0, 6144)
        const unsigned short v = f2bf_bits(acc[mi][ni][r]);
        const int b = m >> 11, t = m & (Tn - 1);
        const int which = n >> 11, rem = n & (Cn - 1);
        const int hh = rem >> 7, d = rem & (Dn - 1);
        const size_t bh = (size_t)(b * Hn + hh);
        if (which == 0)      Qb[(bh * Tn + t) * Dn + d] = v;
        else if (which == 1) Kb[(bh * Tn + t) * Dn + d] = v;
        else                 Vt[(bh * Dn + d) * Tn + t] = v;
      }
}

// ---------------------------------------------------------------------------
// Kernel 2: RoPE in-place on Q and K ([B,H,T,D] bf16). Thread owns (i, i+64).
// ---------------------------------------------------------------------------
__global__ __launch_bounds__(256) void rope_kernel(unsigned short* __restrict__ Qb,
                                                   unsigned short* __restrict__ Kb) {
  const int idx = blockIdx.x * blockDim.x + threadIdx.x;  // B*H*T*64 threads
  const int i  = idx & 63;
  const int t  = (idx >> 6) & (Tn - 1);
  const int bh = idx >> 17;
  // inv_freq = 10000^(-i/64) = exp(-(i/64)*ln(10000))
  const float invf = __expf(-(float)i * (9.210340371976184f / 64.0f));
  const float ang = (float)t * invf;
  const float c = cosf(ang), s = sinf(ang);

  const size_t base = ((size_t)bh * Tn + t) * Dn;
  {
    float q0 = bf2f(Qb[base + i]), q1 = bf2f(Qb[base + i + 64]);
    Qb[base + i]      = f2bf_bits(q0 * c - q1 * s);
    Qb[base + i + 64] = f2bf_bits(q1 * c + q0 * s);
  }
  {
    float k0 = bf2f(Kb[base + i]), k1 = bf2f(Kb[base + i + 64]);
    Kb[base + i]      = f2bf_bits(k0 * c - k1 * s);
    Kb[base + i + 64] = f2bf_bits(k1 * c + k0 * s);
  }
}

// ---------------------------------------------------------------------------
// Kernel 3: causal flash attention. One wave per (b,h, 16-row q-tile).
// k-tiles of 32: S(16x32) = Q(16x128) K^T  -> online softmax -> P via LDS
// transpose into A-fragment -> O(16x128) += P(16x32) V(32x128 via Vt).
// ---------------------------------------------------------------------------
__global__ __launch_bounds__(256) void flash_attn_kernel(
    const unsigned short* __restrict__ Qb, const unsigned short* __restrict__ Kb,
    const unsigned short* __restrict__ Vt, unsigned short* __restrict__ Ob) {
  __shared__ unsigned short Plds[8][16 * 32];    // 1KB per wave

  const int lane = threadIdx.x & 31;
  const int wave = threadIdx.x >> 5;
  const int gw   = blockIdx.x * 8 + wave;
  const int qt = gw & (Tn / 16 - 1);             // 128 q-tiles per (b,h)
  const int bh = gw >> 7;
  const int q0 = qt * 16;
  const int nl = lane & 15, hi = (lane >> 4) & 1;
  const int b = bh >> 4, h = bh & 15;

  const unsigned short* Q = Qb + (size_t)bh * Tn * Dn;
  const unsigned short* K = Kb + (size_t)bh * Tn * Dn;
  const unsigned short* V = Vt + (size_t)bh * Dn * Tn;

  // Q tile A-fragments are loop invariant (16 x 128 = 4 fragments)
  v16bf qa[4];
#pragma unroll
  for (int kc = 0; kc < 4; ++kc) qa[kc] = load_a_bf16(Q, Dn, q0, kc * 32, lane);

  v8f o[8] = {};
  float mrow[8], lrow[8];
#pragma unroll
  for (int r = 0; r < 8; ++r) { mrow[r] = -3.0e38f; lrow[r] = 0.0f; }

  const float scale = 0.08838834764831845f;      // 1/sqrt(128)
  const int ktEnd = (q0 + 16 + 31) >> 5;
  unsigned short* tile = Plds[wave];

  for (int kt = 0; kt < ktEnd; ++kt) {
    // ---- S = Q K^T (two 16x16 accumulators spanning 32 keys) ----
    v8f s0 = {}, s1 = {};
#pragma unroll
    for (int kc = 0; kc < 4; ++kc) {
      v16bf kb0 = load_bT_bf16(K, Dn, kt * 32,      kc * 32, lane);
      v16bf kb1 = load_bT_bf16(K, Dn, kt * 32 + 16, kc * 32, lane);
      s0 = wmma_bf16(qa[kc], kb0, s0);
      s1 = wmma_bf16(qa[kc], kb1, s1);
    }
    // ---- causal mask + online softmax, write P (bf16) to LDS ----
#pragma unroll
    for (int r = 0; r < 8; ++r) {
      const int qrow = q0 + r + 8 * hi;
      const int kg0 = kt * 32 + nl;
      const int kg1 = kg0 + 16;
      const bool in0 = (kg0 <= qrow), in1 = (kg1 <= qrow);
      const float v0 = in0 ? s0[r] * scale : -3.0e38f;
      const float v1 = in1 ? s1[r] * scale : -3.0e38f;
      const float mnew = fmaxf(mrow[r], group_max16(fmaxf(v0, v1)));
      const float alpha = __expf(mrow[r] - mnew);
      mrow[r] = mnew;
      const float p0 = in0 ? __expf(v0 - mnew) : 0.0f;
      const float p1 = in1 ? __expf(v1 - mnew) : 0.0f;
      lrow[r] = lrow[r] * alpha + group_sum16(p0 + p1);
#pragma unroll
      for (int dt = 0; dt < 8; ++dt) o[dt][r] *= alpha;
      tile[(r + 8 * hi) * 32 + nl]      = f2bf_bits(p0);
      tile[(r + 8 * hi) * 32 + 16 + nl] = f2bf_bits(p1);
    }
    // intra-wave LDS transpose: fence compiler + wait DS ops
    __builtin_amdgcn_wave_barrier();
    asm volatile("s_wait_dscnt 0" ::: "memory");
    __builtin_amdgcn_wave_barrier();

    // read P back as an A fragment (16x32); contiguous 16B runs -> ds_load_b128
    v16bf pa;
    {
      const unsigned short* rp = tile + nl * 32;
      const int h8 = 8 * hi;
#pragma unroll
      for (int j = 0; j < 8; ++j) {
        pa[j]     = __builtin_bit_cast(__bf16, rp[h8 + j]);
        pa[8 + j] = __builtin_bit_cast(__bf16, rp[16 + h8 + j]);
      }
    }
    // ---- O += P V  (V transposed -> contiguous B fragments) ----
#pragma unroll
    for (int dt = 0; dt < 8; ++dt) {
      v16bf vb = load_bT_bf16(V, Tn, dt * 16, kt * 32, lane);
      o[dt] = wmma_bf16(pa, vb, o[dt]);
    }
  }

  // ---- normalize, scatter into O buffer [B,T,C] bf16 ----
#pragma unroll
  for (int r = 0; r < 8; ++r) {
    const float inv = 1.0f / lrow[r];
    const int t = q0 + r + 8 * hi;
#pragma unroll
    for (int dt = 0; dt < 8; ++dt) {
      const int c = h * Dn + dt * 16 + nl;
      Ob[((size_t)b * Tn + t) * Cn + c] = f2bf_bits(o[dt][r] * inv);
    }
  }
}

// ---------------------------------------------------------------------------
// Kernel 4: out = O @ w_out^T   (8192 x 2048) x (2048 x 2048)^T, fp32 result
// One wave -> 32x64 tile (2x4 accumulators).
// ---------------------------------------------------------------------------
__global__ __launch_bounds__(256) void out_gemm_kernel(
    const unsigned short* __restrict__ O, const unsigned short* __restrict__ w,
    float* __restrict__ out) {
  const int lane = threadIdx.x & 31;
  const int wave = threadIdx.x >> 5;
  const int gw   = blockIdx.x * 8 + wave;
  const int tilesN = Cn / 64;                    // 32
  const int tm = (gw / tilesN) * 32;
  const int tn = (gw % tilesN) * 64;

  v8f acc[2][4] = {};
  for (int k0 = 0; k0 < Cn; k0 += 32) {
    v16bf a0 = load_a_bf16(O, Cn, tm,      k0, lane);
    v16bf a1 = load_a_bf16(O, Cn, tm + 16, k0, lane);
#pragma unroll
    for (int ni = 0; ni < 4; ++ni) {
      v16bf b = load_bT_bf16(w, Cn, tn + ni * 16, k0, lane);
      acc[0][ni] = wmma_bf16(a0, b, acc[0][ni]);
      acc[1][ni] = wmma_bf16(a1, b, acc[1][ni]);
    }
  }

  const int nl = lane & 15;
  const int hi = (lane >> 4) & 1;
#pragma unroll
  for (int mi = 0; mi < 2; ++mi)
#pragma unroll
    for (int ni = 0; ni < 4; ++ni)
#pragma unroll
      for (int r = 0; r < 8; ++r) {
        const int m = tm + mi * 16 + r + 8 * hi;
        const int n = tn + ni * 16 + nl;
        out[(size_t)m * Cn + n] = acc[mi][ni][r];
      }
}

// ---------------------------------------------------------------------------
// Host entry
// ---------------------------------------------------------------------------
extern "C" void kernel_launch(void* const* d_in, const int* in_sizes, int n_in,
                              void* d_out, int out_size, void* d_ws, size_t ws_size,
                              hipStream_t stream) {
  const float* x     = (const float*)d_in[0];   // [B,T,C]
  const float* w_qkv = (const float*)d_in[1];   // [3C,C]
  const float* w_out = (const float*)d_in[2];   // [C,C]
  float* out = (float*)d_out;                   // [B,T,C] fp32

  char* ws = (char*)d_ws;
  const size_t MB = 1024 * 1024;
  unsigned short* xb  = (unsigned short*)(ws);              // 32 MB
  unsigned short* wqb = (unsigned short*)(ws + 32  * MB);   // 24 MB
  unsigned short* wob = (unsigned short*)(ws + 56  * MB);   //  8 MB
  unsigned short* Qb  = (unsigned short*)(ws + 64  * MB);   // 32 MB
  unsigned short* Kb  = (unsigned short*)(ws + 96  * MB);   // 32 MB
  unsigned short* Vt  = (unsigned short*)(ws + 128 * MB);   // 32 MB
  unsigned short* Ob  = (unsigned short*)(ws + 160 * MB);   // 32 MB

  // 0) one-shot f32 -> bf16 conversions (keeps GEMM hot loops conversion-free)
  const int nx = Mrows * Cn, nwq = NQKV * Cn, nwo = Cn * Cn;
  cvt_bf16_kernel<<<nx  / 8 / 256, 256, 0, stream>>>(x,     (unsigned int*)xb,  nx  / 8);
  cvt_bf16_kernel<<<nwq / 8 / 256, 256, 0, stream>>>(w_qkv, (unsigned int*)wqb, nwq / 8);
  cvt_bf16_kernel<<<nwo / 8 / 256, 256, 0, stream>>>(w_out, (unsigned int*)wob, nwo / 8);

  // 1) QKV projection: (8192/32)*(6144/64)=24576 waves -> 3072 blocks of 8 waves
  qkv_gemm_kernel<<<(Mrows / 32) * (NQKV / 64) / 8, 256, 0, stream>>>(xb, wqb, Qb, Kb, Vt);
  // 2) RoPE: B*H*T*64 threads
  rope_kernel<<<(Bn * Hn * Tn * 64) / 256, 256, 0, stream>>>(Qb, Kb);
  // 3) Flash attention: B*H*(T/16)=8192 waves -> 1024 blocks
  flash_attn_kernel<<<(Bn * Hn * (Tn / 16)) / 8, 256, 0, stream>>>(Qb, Kb, Vt, Ob);
  // 4) Output projection: (8192/32)*(2048/64)=8192 waves -> 1024 blocks
  out_gemm_kernel<<<(Mrows / 32) * (Cn / 64) / 8, 256, 0, stream>>>(Ob, wob, out);
}
